// Attention2_23476291240190
// MI455X (gfx1250) — matrix operand order repaired and checked by reading, not compiled, not run on hardware
//
#include <hip/hip_runtime.h>
#include <hip/hip_bf16.h>

typedef __attribute__((ext_vector_type(16))) _Float16 v16h;
typedef __attribute__((ext_vector_type(8)))  _Float16 v8h;
typedef __attribute__((ext_vector_type(8)))  float    v8f;

#define WMMA16(a, b, c) \
  __builtin_amdgcn_wmma_f32_16x16x32_f16(false, (a), false, (b), (short)0, (c), false, false)

// ---- WMMA fragment loaders (wave32, contiguous k=16*hi+s, same for A and B) ------------
// A (16x32 f16, M=row=lane&15): lane holds K=hi*16..hi*16+15, hi=lane>>4
static __device__ __forceinline__ v16h load_a_rm(const _Float16* p, int ld) {
  int l = threadIdx.x & 31;
  const _Float16* r = p + (size_t)(l & 15) * ld + ((l >> 4) * 16);
  v16h a;
  ((v8h*)&a)[0] = *(const v8h*)(r);
  ((v8h*)&a)[1] = *(const v8h*)(r + 8);
  return a;
}
// B (32x16 f16) from K-contiguous storage: element[K,N] = p[N*ld + K]
// lane: N=lane&15, K group = (lane>>4)*16 .. +15  -> one 32B load
static __device__ __forceinline__ v16h load_b_kt(const _Float16* p, int ld) {
  int l = threadIdx.x & 31;
  return *(const v16h*)(p + (size_t)(l & 15) * ld + ((l >> 4) * 16));
}

// ---- Stage 1: 3x3 conv + bias + BN + ReLU (f32, direct) ---------------------
__global__ __launch_bounds__(256) void conv3x3_bn_relu_kernel(
    const float* __restrict__ x, const float* __restrict__ w,
    const float* __restrict__ bias, const float* __restrict__ bns,
    const float* __restrict__ bnb, float* __restrict__ y, int H, int total) {
  int idx = blockIdx.x * 256 + threadIdx.x;
  if (idx >= total) return;
  int hw = H * H;
  int p  = idx % hw;
  int oc = (idx / hw) & 63;
  int bi = idx / (hw * 64);
  int oy = p / H, ox = p % H;
  float sum = bias[oc];
  const float* wp = w + (size_t)oc * 64 * 9;
  const float* xb = x + (size_t)bi * 64 * hw;
  for (int ic = 0; ic < 64; ++ic) {
    const float* xc = xb + (size_t)ic * hw;
    const float* wc = wp + ic * 9;
#pragma unroll
    for (int ky = 0; ky < 3; ++ky) {
      int yy = oy + ky - 1;
      if ((unsigned)yy >= (unsigned)H) continue;
#pragma unroll
      for (int kx = 0; kx < 3; ++kx) {
        int xx = ox + kx - 1;
        if ((unsigned)xx >= (unsigned)H) continue;
        sum += wc[ky * 3 + kx] * xc[yy * H + xx];
      }
    }
  }
  float v = sum * bns[oc] + bnb[oc];
  y[idx] = v > 0.f ? v : 0.f;
}

// ---- Stage 2: bilinear resize -> feats f32 (c,p) + featsT f16 (p,c) ---------
__global__ __launch_bounds__(256) void resize_kernel(
    const float* __restrict__ src, int H, float* __restrict__ feats,
    _Float16* __restrict__ featsT, int lvl) {
  int idx = blockIdx.x * 256 + threadIdx.x;
  if (idx >= 4 * 64 * 2304) return;
  int p = idx % 2304;
  int c = (idx / 2304) & 63;
  int b = idx / (2304 * 64);
  int oy = p / 48, ox = p % 48;
  float scale = (float)H / 48.0f;
  float sy = (oy + 0.5f) * scale - 0.5f;
  float sx = (ox + 0.5f) * scale - 0.5f;
  sy = fminf(fmaxf(sy, 0.f), (float)(H - 1));
  sx = fminf(fmaxf(sx, 0.f), (float)(H - 1));
  int y0 = (int)sy, x0 = (int)sx;
  int y1 = min(y0 + 1, H - 1), x1 = min(x0 + 1, H - 1);
  float fy = sy - y0, fx = sx - x0;
  const float* sp = src + (size_t)(b * 64 + c) * H * H;
  float v00 = sp[y0 * H + x0], v01 = sp[y0 * H + x1];
  float v10 = sp[y1 * H + x0], v11 = sp[y1 * H + x1];
  float val = (v00 * (1.f - fx) + v01 * fx) * (1.f - fy) +
              (v10 * (1.f - fx) + v11 * fx) * fy;
  feats[((size_t)((lvl * 4 + b) * 64 + c)) * 2304 + p] = val;
  featsT[((size_t)((lvl * 4 + b) * 2304 + p)) * 64 + c] = (_Float16)val;
}

// ---- f32 -> f16 weight conversion -------------------------------------------
__global__ __launch_bounds__(256) void f32_to_f16_kernel(
    const float* __restrict__ in, _Float16* __restrict__ out, int n) {
  int idx = blockIdx.x * 256 + threadIdx.x;
  if (idx < n) out[idx] = (_Float16)in[idx];
}

// ---- Stage 3: QKV GEMM (1536x64)@(64x2304) + BN + ReLU ----------------------
// grid: (432, 1, 28); each wave computes a 16x64 strip.
__global__ __launch_bounds__(256) void qkv_gemm_kernel(
    const _Float16* __restrict__ w16, const _Float16* __restrict__ featsT,
    const float* __restrict__ bns, const float* __restrict__ bnb,
    _Float16* __restrict__ q, _Float16* __restrict__ k,
    _Float16* __restrict__ vT) {
  int ib = blockIdx.z;
  int i = ib >> 2, b = ib & 3;
  int wave = threadIdx.x >> 5;
  int s = blockIdx.x * 8 + wave;          // 0..3455
  int mi = s % 96;                        // M tile (o)
  int n0 = (s / 96) * 64;                 // N strip start (p)
  const _Float16* A  = w16 + (size_t)i * 1536 * 64 + (size_t)mi * 16 * 64;
  const _Float16* fb = featsT + (size_t)(i * 4 + b) * 2304 * 64;
  v8f acc[4] = {};
#pragma unroll
  for (int kk = 0; kk < 64; kk += 32) {
    v16h a = load_a_rm(A + kk, 64);
#pragma unroll
    for (int t = 0; t < 4; ++t) {
      v16h bf = load_b_kt(fb + (size_t)(n0 + t * 16) * 64 + kk, 64);
      acc[t] = WMMA16(a, bf, acc[t]);
    }
  }
  int l = threadIdx.x & 31, ncol = l & 15, hi = l >> 4;
#pragma unroll
  for (int t = 0; t < 4; ++t) {
#pragma unroll
    for (int r = 0; r < 8; ++r) {
      int o = mi * 16 + r + 8 * hi;
      int p = n0 + t * 16 + ncol;
      float v = acc[t][r] * bns[i * 1536 + o] + bnb[i * 1536 + o];
      v = v > 0.f ? v : 0.f;
      int which = o >> 9, rem = o & 511;
      int h = rem >> 6, c64 = rem & 63;
      if (which == 2) {
        vT[((size_t)((i * 4 + b) * 8 + h) * 2304 + p) * 64 + c64] = (_Float16)v;
      } else {
        _Float16* dst = which ? k : q;
        dst[((size_t)((i * 4 + b) * 8 + h) * 64 + c64) * 2304 + p] = (_Float16)v;
      }
    }
  }
}

// ---- Stage 4: attention: dots (async-LDS staged), softmax, sum_j, @v --------
// grid: 224 blocks, one per (i,b,h); 256 threads = 8 waves.
// k panels are staged cooperatively into LDS with GLOBAL_LOAD_ASYNC_TO_LDS_B128
// (double buffered, ASYNCcnt + barrier), then all 8 waves consume them.
#define KCH 128   // K-depth per staged chunk (halves)
#define NCH 18    // 2304 / 128
__global__ __launch_bounds__(256) void attention_kernel(
    const _Float16* __restrict__ q, const _Float16* __restrict__ k,
    const _Float16* __restrict__ vT, _Float16* __restrict__ attoutT) {
  __shared__ float dots[64 * 64];
  __shared__ float asum[64 * 64];
  __shared__ _Float16 kstage[2][64 * KCH];   // 2 x 16 KB
  int blk = blockIdx.x;
  int i = blk >> 5, b = (blk >> 3) & 3, h = blk & 7;
  int tid = threadIdx.x, wave = tid >> 5, l = tid & 31;
  int ncol = l & 15, hi = l >> 4;
  for (int t = tid; t < 4096; t += 256) asum[t] = 0.f;

  int mi = wave >> 1;                 // q row-tile
  int nia = (wave & 1) * 2;           // k col-tiles nia, nia+1
  const _Float16* qp = q + ((size_t)((i * 4 + b) * 8 + h) * 64 + mi * 16) * 2304;
  const unsigned kst0 = (unsigned)(uintptr_t)&kstage[0][0];  // LDS byte offset
  const unsigned kst1 = (unsigned)(uintptr_t)&kstage[1][0];

  for (int j = 0; j < 7; ++j) {
    const _Float16* kp = k + (size_t)((j * 4 + b) * 8 + h) * 64 * 2304;
    // prologue: async-stage chunk 0 into buffer 0 (each thread copies 4x16B)
    __syncthreads();
#pragma unroll
    for (int it = 0; it < 4; ++it) {
      int boff = (it * 256 + tid) * 16;              // byte offset in chunk
      int row  = boff >> 8;                          // 256 B per staged row
      int kb   = (boff & 255) >> 1;                  // half offset within row
      const _Float16* gp = kp + (size_t)row * 2304 + kb;
      unsigned lds = kst0 + (unsigned)boff;
      asm volatile("global_load_async_to_lds_b128 %0, %1, off"
                   :: "v"(lds), "v"(gp) : "memory");
    }
    v8f c0 = {}, c1 = {};
    for (int ch = 0; ch < NCH; ++ch) {
      asm volatile("s_wait_asynccnt 0x0" ::: "memory");
      __syncthreads();                               // chunk ch resident
      int bufp = ch & 1;
      if (ch + 1 < NCH) {                            // stage next chunk
        int kk0n = (ch + 1) * KCH;
        unsigned base = bufp ? kst0 : kst1;
#pragma unroll
        for (int it = 0; it < 4; ++it) {
          int boff = (it * 256 + tid) * 16;
          int row  = boff >> 8;
          int kb   = (boff & 255) >> 1;
          const _Float16* gp = kp + (size_t)row * 2304 + kk0n + kb;
          unsigned lds = base + (unsigned)boff;
          asm volatile("global_load_async_to_lds_b128 %0, %1, off"
                       :: "v"(lds), "v"(gp) : "memory");
        }
      }
      __builtin_prefetch(qp + ch * KCH + 512, 0, 1);
#pragma unroll
      for (int kks = 0; kks < KCH; kks += 32) {
        v16h a = load_a_rm(qp + ch * KCH + kks, 2304);
        v16h b0 = *(const v16h*)&kstage[bufp][(nia * 16 + ncol) * KCH + kks + hi * 16];
        v16h b1 = *(const v16h*)&kstage[bufp][((nia + 1) * 16 + ncol) * KCH + kks + hi * 16];
        c0 = WMMA16(a, b0, c0);
        c1 = WMMA16(a, b1, c1);
      }
      __syncthreads();                               // done reading bufp
    }
#pragma unroll
    for (int r = 0; r < 8; ++r) {
      int row = mi * 16 + r + 8 * hi;
      dots[row * 64 + nia * 16 + ncol]       = c0[r] * 0.125f;  // SCALE = 64^-0.5
      dots[row * 64 + (nia + 1) * 16 + ncol] = c1[r] * 0.125f;
    }
    __syncthreads();
    if (tid < 64) {  // one thread per row c: softmax over e, accumulate sum_j
      float mx = -3.4e38f;
      for (int e = 0; e < 64; ++e) mx = fmaxf(mx, dots[tid * 64 + e]);
      float ssum = 0.f;
      for (int e = 0; e < 64; ++e) {
        float ex = __expf(dots[tid * 64 + e] - mx);
        dots[tid * 64 + e] = ex;
        ssum += ex;
      }
      float inv = 1.f / ssum;
      for (int e = 0; e < 64; ++e) asum[tid * 64 + e] += dots[tid * 64 + e] * inv;
    }
    __syncthreads();
  }

  // out = asum(64x64) @ v(64x2304); A from LDS (f32->f16), B from vT (p,e)
  const _Float16* vp = vT + (size_t)((i * 4 + b) * 8 + h) * 2304 * 64;
  _Float16* ob = attoutT + (size_t)(i * 4 + b) * 2304 * 512;
  for (int t = wave; t < 576; t += 8) {
    int m2 = t & 3, nt = t >> 2;
    v8f c = {};
#pragma unroll
    for (int kk = 0; kk < 64; kk += 32) {
      v16h a;
      int row = m2 * 16 + ncol;
#pragma unroll
      for (int e = 0; e < 16; ++e) {
        int kidx = kk + hi * 16 + e;
        a[e] = (_Float16)asum[row * 64 + kidx];
      }
      v16h bf = load_b_kt(vp + (size_t)(nt * 16) * 64 + kk, 64);
      c = WMMA16(a, bf, c);
    }
#pragma unroll
    for (int r = 0; r < 8; ++r) {
      int p  = nt * 16 + ncol;
      int cc = h * 64 + m2 * 16 + r + 8 * hi;
      ob[(size_t)p * 512 + cc] = (_Float16)c[r];
    }
  }
}

// ---- Stage 5: output projection (64x512)@(512x2304) + BN + ReLU + residual --
// grid: (72, 1, 28); each wave one 16x16 tile, K loop = 512.
__global__ __launch_bounds__(256) void outproj_kernel(
    const _Float16* __restrict__ ow16, const _Float16* __restrict__ attoutT,
    const float* __restrict__ bns, const float* __restrict__ bnb,
    const float* __restrict__ feats, float* __restrict__ out) {
  int ib = blockIdx.z;
  int i = ib >> 2, b = ib & 3;
  int wave = threadIdx.x >> 5;
  int t = blockIdx.x * 8 + wave;   // 0..575
  int mi = t & 3, nt = t >> 2;
  const _Float16* A = ow16 + (size_t)i * 64 * 512 + (size_t)mi * 16 * 512;
  const _Float16* B = attoutT + (size_t)(i * 4 + b) * 2304 * 512 +
                      (size_t)nt * 16 * 512;
  v8f c = {};
  for (int kk = 0; kk < 512; kk += 32) {
    v16h a  = load_a_rm(A + kk, 512);
    v16h bf = load_b_kt(B + kk, 512);
    c = WMMA16(a, bf, c);
  }
  int l = threadIdx.x & 31, ncol = l & 15, hi = l >> 4;
#pragma unroll
  for (int r = 0; r < 8; ++r) {
    int o = mi * 16 + r + 8 * hi;
    int p = nt * 16 + ncol;
    float v = c[r] * bns[i * 64 + o] + bnb[i * 64 + o];
    v = v > 0.f ? v : 0.f;
    size_t idx = ((size_t)((i * 4 + b) * 64 + o)) * 2304 + p;
    out[idx] = v + feats[idx];
  }
}

// ---- Host launch ------------------------------------------------------------
extern "C" void kernel_launch(void* const* d_in, const int* in_sizes, int n_in,
                              void* d_out, int out_size, void* d_ws,
                              size_t ws_size, hipStream_t stream) {
  (void)in_sizes; (void)n_in; (void)out_size; (void)ws_size;
  const float* fin[7];
  for (int i = 0; i < 7; ++i) fin[i] = (const float*)d_in[i];
  const int Hs[7] = {96, 48, 24, 12, 96, 48, 24};
  const float* emb_w    = (const float*)d_in[7];
  const float* emb_b    = (const float*)d_in[8];
  const float* emb_bn_s = (const float*)d_in[9];
  const float* emb_bn_b = (const float*)d_in[10];
  const float* qkv_w    = (const float*)d_in[11];
  const float* qkv_bn_s = (const float*)d_in[12];
  const float* qkv_bn_b = (const float*)d_in[13];
  const float* out_w    = (const float*)d_in[14];
  const float* out_bn_s = (const float*)d_in[15];
  const float* out_bn_b = (const float*)d_in[16];

  const size_t QKV_E   = (size_t)7 * 4 * 8 * 64 * 2304;  // 33,030,144
  const size_t FEATS_E = (size_t)7 * 4 * 64 * 2304;      //  4,128,768

  char* ws = (char*)d_ws;
  size_t off = 0;
  auto alloc = [&](size_t bytes) -> void* {
    void* p = ws + off;
    off = (off + bytes + 255) & ~(size_t)255;
    return p;
  };
  _Float16* attoutT = (_Float16*)alloc(QKV_E * 2);      // 66 MB
  // conv scratch (24.9 MB) overlays attoutT region: dead before attoutT written
  float* convscratch = (float*)attoutT;
  float*     feats   = (float*)alloc(FEATS_E * 4);
  _Float16*  featsT  = (_Float16*)alloc(FEATS_E * 2);
  _Float16*  qkvw16  = (_Float16*)alloc((size_t)7 * 1536 * 64 * 2);
  _Float16*  outw16  = (_Float16*)alloc((size_t)7 * 64 * 512 * 2);
  _Float16*  q16     = (_Float16*)alloc(QKV_E * 2);
  _Float16*  k16     = (_Float16*)alloc(QKV_E * 2);
  _Float16*  vT16    = (_Float16*)alloc(QKV_E * 2);

  // per-level conv scratch offsets
  size_t coff[7];
  {
    size_t o = 0;
    for (int l = 0; l < 7; ++l) { coff[l] = o; o += (size_t)4 * 64 * Hs[l] * Hs[l]; }
  }

  for (int l = 0; l < 7; ++l) {
    int H = Hs[l];
    int total = 4 * 64 * H * H;
    conv3x3_bn_relu_kernel<<<(total + 255) / 256, 256, 0, stream>>>(
        fin[l], emb_w + (size_t)l * 64 * 64 * 9, emb_b + l * 64,
        emb_bn_s + l * 64, emb_bn_b + l * 64, convscratch + coff[l], H, total);
  }
  for (int l = 0; l < 7; ++l) {
    resize_kernel<<<(4 * 64 * 2304 + 255) / 256, 256, 0, stream>>>(
        convscratch + coff[l], Hs[l], feats, featsT, l);
  }
  {
    int n1 = 7 * 1536 * 64;
    f32_to_f16_kernel<<<(n1 + 255) / 256, 256, 0, stream>>>(qkv_w, qkvw16, n1);
    int n2 = 7 * 64 * 512;
    f32_to_f16_kernel<<<(n2 + 255) / 256, 256, 0, stream>>>(out_w, outw16, n2);
  }
  qkv_gemm_kernel<<<dim3(432, 1, 28), 256, 0, stream>>>(
      qkvw16, featsT, qkv_bn_s, qkv_bn_b, q16, k16, vT16);
  attention_kernel<<<224, 256, 0, stream>>>(q16, k16, vT16, attoutT);
  outproj_kernel<<<dim3(72, 1, 28), 256, 0, stream>>>(
      outw16, attoutT, out_bn_s, out_bn_b, feats, (float*)d_out);
}
